// PolarNetDynamicVoxelFeatureExtractor_84774064488766
// MI455X (gfx1250) — matrix operand order repaired and checked by reading, not compiled, not run on hardware
//
#include <hip/hip_runtime.h>

typedef _Float16 v16h __attribute__((ext_vector_type(16)));
typedef _Float16 v8h  __attribute__((ext_vector_type(8)));
typedef float    v8f  __attribute__((ext_vector_type(8)));

#define G0 480
#define G1 360
#define G2 32
#define G01 (G0 * G1)
#define PI_F 3.1415926f
#define PMIN1 (-3.1415926f)
#define PMIN2 (-4.0f)
#define VS0 (50.0f / 480.0f)
#define VS1 ((2.0f * 3.1415926f) / 360.0f)
#define VS2 (6.0f / 32.0f)
#define BN_EPS 1e-5f

__device__ __forceinline__ int clampi(int v, int lo, int hi) {
  return v < lo ? lo : (v > hi ? hi : v);
}

__device__ __forceinline__ v8f wmma16(v16h a, v16h b, v8f c) {
  // D = A(16x32 f16) * B(32x16 f16) + C(16x16 f32)
  return __builtin_amdgcn_wmma_f32_16x16x32_f16(false, a, false, b, (short)0, c,
                                                false, false);
}

// A-fragment loader: lane half hh holds K [hh*16, hh*16+16) of row m
// (contiguous), as two 16B chunks.
__device__ __forceinline__ v16h ldA(const _Float16* p) {
  v8h lo = *(const v8h*)p;
  v8h hi = *(const v8h*)(p + 8);
  v16h r;
#pragma unroll
  for (int i = 0; i < 8; ++i) { r[i] = lo[i]; r[i + 8] = hi[i]; }
  return r;
}

// ---------------------------------------------------------------------------
// Param prep: fold BN into per-channel scale/offset, convert weights to f16.
// BN0 folded into W1 columns + layer-1 offset. Layer outputs: y*s + o (+relu).
// ---------------------------------------------------------------------------
__global__ void prep_params(
    const float* g0, const float* b0, const float* m0, const float* v0,
    const float* g1, const float* bb1, const float* mm1, const float* vv1,
    const float* g2, const float* bb2, const float* mm2, const float* vv2,
    const float* g3, const float* bb3, const float* mm3, const float* vv3,
    const float* W1, const float* lb1, const float* W2, const float* lb2,
    const float* W3, const float* lb3, const float* W4, const float* lb4,
    const float* Wc, const float* lbc,
    _Float16* w1f, _Float16* w2h, _Float16* w3h, _Float16* w4h, _Float16* wch,
    float* s1, float* o1, float* s2, float* o2, float* s3, float* o3,
    float* o4, float* ocp) {
  int idx = blockIdx.x * blockDim.x + threadIdx.x;
  if (idx < 2048) {  // W1 folded with BN0, K padded 14->32
    int n = idx >> 5, k = idx & 31;
    float v = 0.f;
    if (k < 14) {
      float s0k = g0[k] * rsqrtf(v0[k] + BN_EPS);
      v = W1[n * 14 + k] * s0k;
    }
    w1f[idx] = (_Float16)v;
    return;
  }
  idx -= 2048;
  if (idx < 8192) { w2h[idx] = (_Float16)W2[idx]; return; }
  idx -= 8192;
  if (idx < 32768) { w3h[idx] = (_Float16)W3[idx]; return; }
  idx -= 32768;
  if (idx < 65536) { w4h[idx] = (_Float16)W4[idx]; return; }
  idx -= 65536;
  if (idx < 8192) { wch[idx] = (_Float16)Wc[idx]; return; }
  idx -= 8192;
  if (idx < 64) {  // layer1 scale/offset (BN1 + BN0 offset folded through W1)
    int n = idx;
    float s = g1[n] * rsqrtf(vv1[n] + BN_EPS);
    float acc = lb1[n];
#pragma unroll
    for (int k = 0; k < 14; ++k) {
      float s0k = g0[k] * rsqrtf(v0[k] + BN_EPS);
      float o0k = b0[k] - m0[k] * s0k;
      acc += o0k * W1[n * 14 + k];
    }
    s1[n] = s;
    o1[n] = (acc - mm1[n]) * s + bb1[n];
    return;
  }
  idx -= 64;
  if (idx < 128) {
    int n = idx;
    float s = g2[n] * rsqrtf(vv2[n] + BN_EPS);
    s2[n] = s;
    o2[n] = (lb2[n] - mm2[n]) * s + bb2[n];
    return;
  }
  idx -= 128;
  if (idx < 256) {
    int n = idx;
    float s = g3[n] * rsqrtf(vv3[n] + BN_EPS);
    s3[n] = s;
    o3[n] = (lb3[n] - mm3[n]) * s + bb3[n];
    return;
  }
  idx -= 256;
  if (idx < 256) { o4[idx] = lb4[idx]; return; }
  idx -= 256;
  if (idx < 32) { ocp[idx] = lbc[idx]; return; }
}

// ---------------------------------------------------------------------------
// Pass 1: voxelize, per-segment count + 5-feature sums (atomics to L2).
// ---------------------------------------------------------------------------
__global__ void k_points(const float* __restrict__ pts, int N,
                         float* __restrict__ cnt, float* __restrict__ sum5,
                         int* __restrict__ segbuf, float* __restrict__ rhophi) {
  int i = blockIdx.x * blockDim.x + threadIdx.x;
  if (i >= N) return;
  float bf = pts[i * 5 + 0];
  float x = pts[i * 5 + 1], y = pts[i * 5 + 2], z = pts[i * 5 + 3];
  float rho = sqrtf(x * x + y * y);
  float phi = atan2f(y, x);
  int v0 = clampi((int)floorf(rho * (1.0f / VS0)), 0, G0 - 1);
  int v1 = clampi((int)floorf((phi - PMIN1) * (1.0f / VS1)), 0, G1 - 1);
  int b = (int)bf;
  int seg = (b * G0 + v0) * G1 + v1;
  segbuf[i] = seg;
  rhophi[2 * i] = rho;
  rhophi[2 * i + 1] = phi;
  atomicAdd(cnt + seg, 1.0f);
  float f5[5] = {rho, phi, z, x, y};
#pragma unroll
  for (int j = 0; j < 5; ++j) atomicAdd(sum5 + (size_t)seg * 5 + j, f5[j]);
}

// ---------------------------------------------------------------------------
// Hidden layer: D(lds,16xDIN f16) -> WMMA -> scale/offset(+relu) -> back to
// LDS in A layout (act row stride fixed at 256 halves).
// ---------------------------------------------------------------------------
template <int DIN, int DOUT, bool RELU>
__device__ __forceinline__ void mlp_layer(const _Float16* __restrict__ W,
                                          const float* __restrict__ S,
                                          const float* __restrict__ O,
                                          _Float16* act, int lane) {
  constexpr int KS = DIN / 32;
  const int m = lane & 15, hh = lane >> 4;
  v16h a[KS];
#pragma unroll
  for (int kk = 0; kk < KS; ++kk)
    a[kk] = ldA(&act[m * 256 + kk * 32 + hh * 16]);
#pragma unroll
  for (int nt = 0; nt < DOUT / 16; ++nt) {
    v8f c = {0.f, 0.f, 0.f, 0.f, 0.f, 0.f, 0.f, 0.f};
#pragma unroll
    for (int kk = 0; kk < KS; ++kk) {
      // B layout: lane n holds column n = W row (nt*16+n); lane half hh
      // covers K window [kk*32 + hh*16, +16) — 32B contiguous.
      v16h b = *(const v16h*)(W + (nt * 16 + m) * DIN + kk * 32 + hh * 16);
      c = wmma16(a[kk], b, c);
    }
    const int ch = nt * 16 + m;
    const float sc = S[ch], of = O[ch];
#pragma unroll
    for (int j = 0; j < 8; ++j) {  // D: row = hh*8+j, col = ch
      float v = c[j] * sc + of;
      if (RELU) v = fmaxf(v, 0.f);
      act[(hh * 8 + j) * 256 + ch] = (_Float16)v;
    }
  }
}

// Final layer (256->256, bias only): accumulate into pool via f32 atomics.
template <int DIN>
__device__ __forceinline__ void mlp_final(const _Float16* __restrict__ W,
                                          const float* __restrict__ O,
                                          const _Float16* act,
                                          const int* segRow,
                                          float* __restrict__ pool, int lane,
                                          bool active) {
  constexpr int KS = DIN / 32;
  const int m = lane & 15, hh = lane >> 4;
  v16h a[KS];
#pragma unroll
  for (int kk = 0; kk < KS; ++kk)
    a[kk] = ldA(&act[m * 256 + kk * 32 + hh * 16]);
#pragma unroll
  for (int nt = 0; nt < 16; ++nt) {
    v8f c = {0.f, 0.f, 0.f, 0.f, 0.f, 0.f, 0.f, 0.f};
#pragma unroll
    for (int kk = 0; kk < KS; ++kk) {
      v16h b = *(const v16h*)(W + (nt * 16 + m) * DIN + kk * 32 + hh * 16);
      c = wmma16(a[kk], b, c);
    }
    const int ch = nt * 16 + m;
    const float of = O[ch];
    if (active) {
#pragma unroll
      for (int j = 0; j < 8; ++j) {
        int mr = hh * 8 + j;
        atomicAdd(&pool[(size_t)segRow[mr] * 256 + ch], c[j] + of);
      }
    }
  }
}

// ---------------------------------------------------------------------------
// Pass 2: one wave = 16 points through the whole MLP (212 WMMAs / tile).
// ---------------------------------------------------------------------------
#define WPB 4
__global__ void __launch_bounds__(32 * WPB) k_mlp(
    const float* __restrict__ pts, const int* __restrict__ segbuf,
    const float* __restrict__ rhophi, const float* __restrict__ cnt,
    const float* __restrict__ sum5, int N,
    const _Float16* __restrict__ w1f, const _Float16* __restrict__ w2h,
    const _Float16* __restrict__ w3h, const _Float16* __restrict__ w4h,
    const float* __restrict__ s1, const float* __restrict__ o1,
    const float* __restrict__ s2, const float* __restrict__ o2,
    const float* __restrict__ s3, const float* __restrict__ o3,
    const float* __restrict__ o4, float* __restrict__ pool) {
  __shared__ __align__(32) _Float16 sAct[WPB][16 * 256];
  __shared__ int sSeg[WPB][16];
  const int lane = threadIdx.x & 31, wid = threadIdx.x >> 5;
  const int base = (blockIdx.x * WPB + wid) * 16;
  const bool active = base < N;
  _Float16* act = sAct[wid];

  // Phase A: lanes 0..15 build one 14-dim point feature row each (f16).
  if (lane < 16) {
    _Float16* row = &act[lane * 256];
    if (active) {
      int p = base + lane;
      float x = pts[p * 5 + 1], y = pts[p * 5 + 2];
      float z = pts[p * 5 + 3], inten = pts[p * 5 + 4];
      float rho = rhophi[2 * p], phi = rhophi[2 * p + 1];
      int seg = segbuf[p];
      float cct = cnt[seg];
      float inv = 1.0f / fmaxf(cct, 1.0f);
      float mn0 = sum5[(size_t)seg * 5 + 0] * inv;
      float mn1 = sum5[(size_t)seg * 5 + 1] * inv;
      float mn2 = sum5[(size_t)seg * 5 + 2] * inv;
      float mn3 = sum5[(size_t)seg * 5 + 3] * inv;
      float mn4 = sum5[(size_t)seg * 5 + 4] * inv;
      int v0 = clampi((int)floorf(rho * (1.0f / VS0)), 0, G0 - 1);
      int v1 = clampi((int)floorf((phi - PMIN1) * (1.0f / VS1)), 0, G1 - 1);
      int v2 = clampi((int)floorf((z - PMIN2) * (1.0f / VS2)), 0, G2 - 1);
      float c0 = (v0 + 0.5f) * VS0;
      float c1 = (v1 + 0.5f) * VS1 + PMIN1;
      float c2 = (v2 + 0.5f) * VS2 + PMIN2;
      float f[14] = {rho, phi, z, x, y, inten,
                     rho - mn0, phi - mn1, z - mn2, x - mn3, y - mn4,
                     rho - c0, phi - c1, z - c2};
#pragma unroll
      for (int k = 0; k < 14; ++k) row[k] = (_Float16)f[k];
#pragma unroll
      for (int k = 14; k < 32; ++k) row[k] = (_Float16)0.0f;
      sSeg[wid][lane] = seg;
    } else {
#pragma unroll
      for (int k = 0; k < 32; ++k) row[k] = (_Float16)0.0f;
      sSeg[wid][lane] = 0;
    }
  }
  __syncthreads();
  mlp_layer<32, 64, true>(w1f, s1, o1, act, lane);
  __syncthreads();
  mlp_layer<64, 128, true>(w2h, s2, o2, act, lane);
  __syncthreads();
  mlp_layer<128, 256, true>(w3h, s3, o3, act, lane);
  __syncthreads();
  mlp_final<256>(w4h, o4, act, sSeg[wid], pool, lane, active);
}

// ---------------------------------------------------------------------------
// Pass 3: one wave = 16 segments. mean -> 256x32 WMMA head -> relu/mask ->
// scatter into BEV (b, 32, 480, 360).
// ---------------------------------------------------------------------------
__global__ void __launch_bounds__(32 * WPB) k_head(
    const float* __restrict__ pool, const float* __restrict__ cnt,
    const _Float16* __restrict__ wch, const float* __restrict__ ocp,
    float* __restrict__ out, int NSEG) {
  __shared__ __align__(32) _Float16 sAct[WPB][16 * 256];
  __shared__ float sCnt[WPB][16];
  __shared__ int sBase[WPB][16];
  const int lane = threadIdx.x & 31, wid = threadIdx.x >> 5;
  const int base = (blockIdx.x * WPB + wid) * 16;
  const bool active = base < NSEG;
  _Float16* act = sAct[wid];

  if (active) {  // cooperative: 2 lanes per segment row, 128 floats each
    int m = lane >> 1;
    int hoff = (lane & 1) * 128;
    int s = base + m;
    float cct = cnt[s];
    float inv = 1.0f / fmaxf(cct, 1.0f);
    const float4* src = (const float4*)(pool + (size_t)s * 256 + hoff);
    _Float16* dst = &act[m * 256 + hoff];
#pragma unroll
    for (int i = 0; i < 32; ++i) {
      float4 v = src[i];
      dst[i * 4 + 0] = (_Float16)(v.x * inv);
      dst[i * 4 + 1] = (_Float16)(v.y * inv);
      dst[i * 4 + 2] = (_Float16)(v.z * inv);
      dst[i * 4 + 3] = (_Float16)(v.w * inv);
    }
    if ((lane & 1) == 0) {
      sCnt[wid][m] = cct;
      int b = s / G01;
      int r = s - b * G01;
      int r0 = r / G1;
      int r1 = r - r0 * G1;
      sBase[wid][m] = ((b * 32) * G0 + r0) * G1 + r1;
    }
  }
  __syncthreads();

  const int m = lane & 15, hh = lane >> 4;
  v16h a[8];
#pragma unroll
  for (int kk = 0; kk < 8; ++kk)
    a[kk] = ldA(&act[m * 256 + kk * 32 + hh * 16]);
#pragma unroll
  for (int nt = 0; nt < 2; ++nt) {
    v8f c = {0.f, 0.f, 0.f, 0.f, 0.f, 0.f, 0.f, 0.f};
#pragma unroll
    for (int kk = 0; kk < 8; ++kk) {
      v16h b = *(const v16h*)(wch + (nt * 16 + m) * 256 + kk * 32 + hh * 16);
      c = wmma16(a[kk], b, c);
    }
    const int ch = nt * 16 + m;
    const float of = ocp[ch];
    if (active) {
#pragma unroll
      for (int j = 0; j < 8; ++j) {
        int mr = hh * 8 + j;
        float v = fmaxf(c[j] + of, 0.f);
        if (sCnt[wid][mr] <= 0.f) v = 0.f;
        out[(size_t)sBase[wid][mr] + (size_t)ch * G01] = v;
      }
    }
  }
}

// ---------------------------------------------------------------------------
extern "C" void kernel_launch(void* const* d_in, const int* in_sizes, int n_in,
                              void* d_out, int out_size, void* d_ws,
                              size_t ws_size, hipStream_t stream) {
  const float* pts = (const float*)d_in[0];
  const int NPTS = in_sizes[0] / 5;
  const int B = out_size / (32 * G0 * G1);
  const int NSEG = B * G0 * G1;

  // Workspace carve (256B aligned slots).
  char* w = (char*)d_ws;
  size_t off = 0;
  auto carve = [&](size_t bytes) -> void* {
    void* p = w + off;
    off = (off + bytes + 255) & ~(size_t)255;
    return p;
  };
  float* cnt = (float*)carve((size_t)NSEG * 4);
  float* sum5 = (float*)carve((size_t)NSEG * 5 * 4);
  float* pool = (float*)carve((size_t)NSEG * 256 * 4);
  size_t zeroBytes = off;
  int* segbuf = (int*)carve((size_t)NPTS * 4);
  float* rhophi = (float*)carve((size_t)NPTS * 8);
  _Float16* w1f = (_Float16*)carve(2048 * 2);
  _Float16* w2h = (_Float16*)carve(8192 * 2);
  _Float16* w3h = (_Float16*)carve(32768 * 2);
  _Float16* w4h = (_Float16*)carve(65536 * 2);
  _Float16* wch = (_Float16*)carve(8192 * 2);
  float* s1 = (float*)carve(64 * 4);
  float* o1 = (float*)carve(64 * 4);
  float* s2 = (float*)carve(128 * 4);
  float* o2 = (float*)carve(128 * 4);
  float* s3 = (float*)carve(256 * 4);
  float* o3 = (float*)carve(256 * 4);
  float* o4 = (float*)carve(256 * 4);
  float* ocp = (float*)carve(32 * 4);
  if (ws_size < off) return;  // workspace too small; bail deterministically

  // Param pointers (setup_inputs dict order: points, batch_size, params
  // {bns[4 x (g,b,m,v)], lins[4 x (W,b)], comp (W,b)}).
  const float *g0 = (const float*)d_in[2], *b0 = (const float*)d_in[3],
              *m0 = (const float*)d_in[4], *v0 = (const float*)d_in[5];
  const float *g1 = (const float*)d_in[6], *bb1 = (const float*)d_in[7],
              *mm1 = (const float*)d_in[8], *vv1 = (const float*)d_in[9];
  const float *g2 = (const float*)d_in[10], *bb2 = (const float*)d_in[11],
              *mm2 = (const float*)d_in[12], *vv2 = (const float*)d_in[13];
  const float *g3 = (const float*)d_in[14], *bb3 = (const float*)d_in[15],
              *mm3 = (const float*)d_in[16], *vv3 = (const float*)d_in[17];
  const float *W1 = (const float*)d_in[18], *lb1 = (const float*)d_in[19];
  const float *W2 = (const float*)d_in[20], *lb2 = (const float*)d_in[21];
  const float *W3 = (const float*)d_in[22], *lb3 = (const float*)d_in[23];
  const float *W4 = (const float*)d_in[24], *lb4 = (const float*)d_in[25];
  const float *Wc = (const float*)d_in[26], *lbc = (const float*)d_in[27];

  hipMemsetAsync(d_ws, 0, zeroBytes, stream);

  prep_params<<<459, 256, 0, stream>>>(
      g0, b0, m0, v0, g1, bb1, mm1, vv1, g2, bb2, mm2, vv2, g3, bb3, mm3, vv3,
      W1, lb1, W2, lb2, W3, lb3, W4, lb4, Wc, lbc, w1f, w2h, w3h, w4h, wch, s1,
      o1, s2, o2, s3, o3, o4, ocp);

  k_points<<<(NPTS + 255) / 256, 256, 0, stream>>>(pts, NPTS, cnt, sum5,
                                                   segbuf, rhophi);

  int tiles = (NPTS + 15) / 16;
  k_mlp<<<(tiles + WPB - 1) / WPB, 32 * WPB, 0, stream>>>(
      pts, segbuf, rhophi, cnt, sum5, NPTS, w1f, w2h, w3h, w4h, s1, o1, s2, o2,
      s3, o3, o4, pool);

  int stiles = (NSEG + 15) / 16;
  k_head<<<(stiles + WPB - 1) / WPB, 32 * WPB, 0, stream>>>(
      pool, cnt, wch, ocp, (float*)d_out, NSEG);
}
